// Attention_17824114278463
// MI455X (gfx1250) — matrix operand order repaired and checked by reading, not compiled, not run on hardware
//
#include <hip/hip_runtime.h>

// ---------------------------------------------------------------------------
// Types / helpers for CDNA5 WMMA (wave32, 16x16x32 bf16 -> f32)
// ---------------------------------------------------------------------------
typedef __attribute__((ext_vector_type(16))) __bf16 bf16x16;
typedef __attribute__((ext_vector_type(8)))  float  floatx8;

union BFrag {
    bf16x16      v;
    unsigned int u[8];
};

__device__ __forceinline__ unsigned short f2bf(float f) {
    unsigned int u = __float_as_uint(f);
    u += 0x7FFFu + ((u >> 16) & 1u);   // round-to-nearest-even
    return (unsigned short)(u >> 16);
}
__device__ __forceinline__ unsigned int pack2bf(float lo, float hi) {
    return (unsigned int)f2bf(lo) | ((unsigned int)f2bf(hi) << 16);
}

// ---------------------------------------------------------------------------
// gfx1250 async global->LDS copy (ASYNCcnt-tracked), with sync fallback.
// Builtin signature (from hipcc diagnostic): param0 = int4 (vector_size 16)
// pointer in the global (AS1/__device__) space, param1 = LDS (AS3) int4
// pointer, then imm offset + imm cpol.
// ---------------------------------------------------------------------------
#if defined(__has_builtin)
#if __has_builtin(__builtin_amdgcn_global_load_async_to_lds_b128) && \
    __has_builtin(__builtin_amdgcn_s_wait_asynccnt)
#define USE_ASYNC_LDS 1
#endif
#endif

typedef int v4i_vs __attribute__((vector_size(16)));
typedef __attribute__((address_space(1))) v4i_vs gv4i;   // global int4
typedef __attribute__((address_space(3))) v4i_vs lv4i;   // LDS int4

__device__ __forceinline__ void cp16(const void* g, void* l) {
#ifdef USE_ASYNC_LDS
    __builtin_amdgcn_global_load_async_to_lds_b128(
        (gv4i*)(unsigned long long)g,
        (lv4i*)(unsigned)(unsigned long long)l,
        0, 0);
#else
    *(uint4*)l = *(const uint4*)g;
#endif
}
template <int N> __device__ __forceinline__ void async_wait() {
#ifdef USE_ASYNC_LDS
    __builtin_amdgcn_s_wait_asynccnt(N);
#endif
}

// ---------------------------------------------------------------------------
// im2col: x[64,1,256,256] f32 -> A0[16384,256] bf16 (row = b*256+py*16+px,
// col = ph*16+pw) so patch-embed becomes a plain GEMM.
// ---------------------------------------------------------------------------
__global__ void k_im2col(const float* __restrict__ x, unsigned short* __restrict__ A0) {
    int r   = blockIdx.x;
    int col = threadIdx.x;
    int b  = r >> 8;
    int t  = r & 255;
    int py = t >> 4, px = t & 15;
    int ph = col >> 4, pw = col & 15;
    float v = x[((size_t)b * 256 + py * 16 + ph) * 256 + px * 16 + pw];
    A0[(size_t)r * 256 + col] = f2bf(v);
}

// flat f32 -> bf16 (conv_w [1024,256] is already [N][K] for the patch GEMM)
__global__ void k_cvt(const float* __restrict__ in, unsigned short* __restrict__ out, int n) {
    int i = blockIdx.x * 256 + threadIdx.x;
    if (i < n) out[i] = f2bf(in[i]);
}

// 1024x1024 transpose + convert: out[n][k] = bf16(in[k][n])  (LDS-tiled)
__global__ void k_wt(const float* __restrict__ in, unsigned short* __restrict__ out) {
    __shared__ float tile[32][33];
    int bx = blockIdx.x, by = blockIdx.y;
    int tx = threadIdx.x, ty = threadIdx.y;   // 32 x 8
    #pragma unroll
    for (int j = 0; j < 4; ++j)
        tile[ty + j * 8][tx] = in[(size_t)(by * 32 + ty + j * 8) * 1024 + bx * 32 + tx];
    __syncthreads();
    #pragma unroll
    for (int j = 0; j < 4; ++j)
        out[(size_t)(bx * 32 + ty + j * 8) * 1024 + by * 32 + tx] =
            f2bf(tile[tx][ty + j * 8]);
}

// ---------------------------------------------------------------------------
// bf16 WMMA GEMM: C[M,N] = A[M,K] @ Bt[N,K]^T + bias
//   block tile 128x128, 8 waves of 32x64; async double-buffered LDS staging.
//   fmode: 0 none, 1 flat f32 [M,N], 2 present permute [B,H,T,D]
//   outBf: optional flat bf16 [M,N]
// ---------------------------------------------------------------------------
#define APITCH 40   // ushorts per LDS row: 32 data + 8 pad (80B, 16B aligned)

__launch_bounds__(256)
__global__ void k_gemm(const unsigned short* __restrict__ A,
                       const unsigned short* __restrict__ Bt,
                       const float* __restrict__ bias,
                       float* __restrict__ outF, int fmode,
                       unsigned short* __restrict__ outBf,
                       int M, int N, int K)
{
    __shared__ __align__(16) unsigned short Alds[2][128 * APITCH];
    __shared__ __align__(16) unsigned short Blds[2][128 * APITCH];

    const int tid  = threadIdx.x;
    const int lane = tid & 31;
    const int wv   = tid >> 5;
    const int m0   = blockIdx.y * 128;
    const int n0   = blockIdx.x * 128;
    const int wm   = (wv >> 1) * 32;
    const int wn   = (wv & 1) * 64;
    const int hi   = lane >> 4;

    // Stage one 128x32 A tile + 128x32 B tile (pure 16B copies, 4 per thread).
    auto stage = [&](int kt, int b) {
        const int k0 = kt << 5;
        #pragma unroll
        for (int i = 0; i < 2; ++i) {
            int c   = tid + (i << 8);        // chunk 0..511
            int row = c >> 2;
            int ch  = (c & 3) << 3;          // element offset within row
            cp16(A  + (size_t)(m0 + row) * K + k0 + ch, &Alds[b][row * APITCH + ch]);
            cp16(Bt + (size_t)(n0 + row) * K + k0 + ch, &Blds[b][row * APITCH + ch]);
        }
    };

    floatx8 acc[2][4] = {};
    const int nK = K >> 5;

    stage(0, 0);
    for (int kt = 0; kt < nK; ++kt) {
        const int cur = kt & 1;
        if (kt + 1 < nK) {
            stage(kt + 1, cur ^ 1);
            async_wait<4>();     // current tile landed; next tile in flight
        } else {
            async_wait<0>();
        }
        if (kt + 2 < nK) {       // L2 prefetch of the k+2 tiles
            __builtin_prefetch(A  + (size_t)(m0 + (tid >> 1)) * K + ((kt + 2) << 5), 0, 0);
            __builtin_prefetch(Bt + (size_t)(n0 + (tid >> 1)) * K + ((kt + 2) << 5), 0, 0);
        }
        __syncthreads();

        const unsigned int* A32 = (const unsigned int*)Alds[cur];
        const unsigned int* B32 = (const unsigned int*)Blds[cur];
        BFrag af[2], bf[4];
        #pragma unroll
        for (int tm = 0; tm < 2; ++tm) {
            int m = wm + tm * 16 + (lane & 15);
            #pragma unroll
            for (int v = 0; v < 8; ++v) {
                int kp = ((v >> 2) << 3) + (hi << 2) + (v & 3);  // pair idx 0..15
                af[tm].u[v] = A32[m * (APITCH / 2) + kp];
            }
        }
        #pragma unroll
        for (int tn = 0; tn < 4; ++tn) {
            int n = wn + tn * 16 + (lane & 15);
            #pragma unroll
            for (int v = 0; v < 8; ++v)
                bf[tn].u[v] = B32[n * (APITCH / 2) + ((v >> 2) << 3) + (hi << 2) + (v & 3)];
        }
        #pragma unroll
        for (int tm = 0; tm < 2; ++tm)
            #pragma unroll
            for (int tn = 0; tn < 4; ++tn)
                acc[tm][tn] = __builtin_amdgcn_wmma_f32_16x16x32_bf16(
                    false, af[tm].v, false, bf[tn].v, (short)0, acc[tm][tn],
                    false, false);
        __syncthreads();
    }

    // Epilogue: bias + stores. C layout: VGPR r -> row r + 8*hi, col lane&15.
    const int rhalf = hi << 3;
    for (int tm = 0; tm < 2; ++tm) {
        for (int tn = 0; tn < 4; ++tn) {
            int nn   = n0 + wn + tn * 16 + (lane & 15);
            float bv = bias ? bias[nn] : 0.0f;
            for (int r = 0; r < 8; ++r) {
                int mm    = m0 + wm + tm * 16 + r + rhalf;
                float val = acc[tm][tn][r] + bv;
                if (fmode == 1) {
                    outF[(size_t)mm * N + nn] = val;
                } else if (fmode == 2) {
                    int b = mm >> 8, t = mm & 255, h = nn >> 7, d = nn & 127;
                    outF[(((size_t)(b * 8 + h)) << 15) + ((size_t)t << 7) + d] = val;
                }
                if (outBf) outBf[(size_t)mm * N + nn] = f2bf(val);
            }
        }
    }
}

// ---------------------------------------------------------------------------
// Flash-style causal attention, one (b,h) per 512-thread block (16 waves,
// 16 query rows each). K row-major / V transposed staged as bf16 in LDS with
// bank-conflict-free pitches; online softmax; QK^T and PV on WMMA.
// ---------------------------------------------------------------------------
#define KP 136   // K LDS pitch (ushorts): 128 + 8
#define VP 264   // V LDS pitch (ushorts): 256 + 8
#define PP 40    // P LDS pitch (ushorts): 32 + 8

__launch_bounds__(512)
__global__ void k_attn(const float* __restrict__ Kp,           // [B*H,256,128] f32
                       const float* __restrict__ Vp,           // [B*H,256,128] f32
                       const unsigned short* __restrict__ Qbf, // [16384,1024] bf16
                       unsigned short* __restrict__ Ybf)       // [16384,1024] bf16
{
    extern __shared__ unsigned short smem[];
    unsigned short* Klds = smem;                  // 256 x KP
    unsigned short* Vlds = smem + 256 * KP;       // 128 x VP  (transposed [d][t])
    unsigned short* Plds = Vlds + 128 * VP;       // 16 waves x 16 x PP

    const int bh = blockIdx.x;        // b*8 + h
    const int b  = bh >> 3;
    const int h  = bh & 7;
    const int tid = threadIdx.x, lane = tid & 31, wv = tid >> 5;

    // Cooperative staging: float4 loads, bf16-pack stores.
    const float4* k4 = (const float4*)(Kp + ((size_t)bh << 15));
    const float4* v4 = (const float4*)(Vp + ((size_t)bh << 15));
    #pragma unroll
    for (int i = 0; i < 16; ++i) {
        int fid = tid + (i << 9);                 // 0..8191 float4s
        int t = fid >> 5, d0 = (fid & 31) << 2;
        float4 kk = k4[fid];
        *(unsigned int*)(Klds + t * KP + d0)     = pack2bf(kk.x, kk.y);
        *(unsigned int*)(Klds + t * KP + d0 + 2) = pack2bf(kk.z, kk.w);
        float4 vv = v4[fid];
        Vlds[(d0    ) * VP + t] = f2bf(vv.x);
        Vlds[(d0 + 1) * VP + t] = f2bf(vv.y);
        Vlds[(d0 + 2) * VP + t] = f2bf(vv.z);
        Vlds[(d0 + 3) * VP + t] = f2bf(vv.w);
    }
    __syncthreads();

    const int hi    = lane >> 4;
    const int rhalf = hi << 3;
    const int colh  = lane & 15;
    const int qrow0 = wv * 16;

    // Q fragments: 16 rows x 128 d, 4 K-slices of 32 (direct global b128s).
    BFrag qf[4];
    {
        size_t rowbase = ((size_t)(b * 256 + qrow0 + colh)) * 1024 + h * 128;
        #pragma unroll
        for (int kk = 0; kk < 4; ++kk)
            #pragma unroll
            for (int v = 0; v < 8; ++v) {
                int koff = kk * 32 + ((v >> 2) << 4) + (hi << 3) + ((v & 3) << 1);
                qf[kk].u[v] = *(const unsigned int*)(Qbf + rowbase + koff);
            }
    }

    floatx8 acc[8] = {};
    float mrun[8], lrun[8];
    for (int r = 0; r < 8; ++r) { mrun[r] = -1e30f; lrun[r] = 0.0f; }
    const float scale = 0.08838834764831845f;   // 1/sqrt(128)
    const int qmax = qrow0 + 15;

    const unsigned int* K32 = (const unsigned int*)Klds;
    const unsigned int* V32 = (const unsigned int*)Vlds;
    unsigned short* pw = Plds + wv * 16 * PP;

    for (int kstart = 0; kstart <= qmax; kstart += 32) {
        // S = Q K^T for two 16-key tiles
        floatx8 s[2];
        #pragma unroll
        for (int j = 0; j < 2; ++j) {
            floatx8 c = {};
            int key = kstart + j * 16 + colh;
            #pragma unroll
            for (int kk = 0; kk < 4; ++kk) {
                BFrag bfr;
                #pragma unroll
                for (int v = 0; v < 8; ++v)
                    bfr.u[v] = K32[key * (KP / 2) + kk * 16 + ((v >> 2) << 3) + (hi << 2) + (v & 3)];
                c = __builtin_amdgcn_wmma_f32_16x16x32_bf16(
                        false, qf[kk].v, false, bfr.v, (short)0, c, false, false);
            }
            s[j] = c;
        }

        // Online softmax per query row (16-lane-half reductions).
        #pragma unroll
        for (int r = 0; r < 8; ++r) {
            int qrow = qrow0 + r + rhalf;
            float s0 = s[0][r] * scale;
            float s1 = s[1][r] * scale;
            if (kstart + colh      > qrow) s0 = -1e30f;
            if (kstart + 16 + colh > qrow) s1 = -1e30f;
            float mx = fmaxf(s0, s1);
            mx = fmaxf(mx, __shfl_xor(mx, 1, 32));
            mx = fmaxf(mx, __shfl_xor(mx, 2, 32));
            mx = fmaxf(mx, __shfl_xor(mx, 4, 32));
            mx = fmaxf(mx, __shfl_xor(mx, 8, 32));
            float mnew = fmaxf(mrun[r], mx);
            float corr = __expf(mrun[r] - mnew);
            float p0 = __expf(s0 - mnew);
            float p1 = __expf(s1 - mnew);
            float sum = p0 + p1;
            sum += __shfl_xor(sum, 1, 32);
            sum += __shfl_xor(sum, 2, 32);
            sum += __shfl_xor(sum, 4, 32);
            sum += __shfl_xor(sum, 8, 32);
            lrun[r] = lrun[r] * corr + sum;
            mrun[r] = mnew;
            #pragma unroll
            for (int tn = 0; tn < 8; ++tn) acc[tn][r] *= corr;
            int row = r + rhalf;
            pw[row * PP + colh]      = f2bf(p0);
            pw[row * PP + 16 + colh] = f2bf(p1);
        }
        // Per-wave LDS round-trip: order the P stores before fragment reads.
        asm volatile("s_wait_dscnt 0" ::: "memory");

        // p -> A fragment (16x32 bf16)
        BFrag pf;
        {
            const unsigned int* P32 = (const unsigned int*)pw;
            #pragma unroll
            for (int v = 0; v < 8; ++v) {
                int kp = ((v >> 2) << 3) + (hi << 2) + (v & 3);
                pf.u[v] = P32[colh * (PP / 2) + kp];
            }
        }
        // acc += p @ V(32 keys x 128 d)
        #pragma unroll
        for (int tn = 0; tn < 8; ++tn) {
            int n = tn * 16 + colh;
            BFrag bfr;
            #pragma unroll
            for (int v = 0; v < 8; ++v)
                bfr.u[v] = V32[n * (VP / 2) + (kstart >> 1) + ((v >> 2) << 3) + (hi << 2) + (v & 3)];
            acc[tn] = __builtin_amdgcn_wmma_f32_16x16x32_bf16(
                          false, pf.v, false, bfr.v, (short)0, acc[tn], false, false);
        }
    }

    // Normalize and write Y (bf16, [b*256+t][h*128+d]) for the output proj.
    for (int r = 0; r < 8; ++r) {
        int qrow  = qrow0 + r + rhalf;
        float inv = 1.0f / lrun[r];
        size_t base = ((size_t)(b * 256 + qrow)) * 1024 + h * 128;
        for (int tn = 0; tn < 8; ++tn)
            Ybf[base + tn * 16 + colh] = f2bf(acc[tn][r] * inv);
    }
}

// ---------------------------------------------------------------------------
// Host-side orchestration.
// ---------------------------------------------------------------------------
extern "C" void kernel_launch(void* const* d_in, const int* in_sizes, int n_in,
                              void* d_out, int out_size, void* d_ws, size_t ws_size,
                              hipStream_t stream) {
    const float* x      = (const float*)d_in[0];
    const float* conv_w = (const float*)d_in[1];
    const float* conv_b = (const float*)d_in[2];
    const float* Wq     = (const float*)d_in[3];
    const float* bq     = (const float*)d_in[4];
    const float* Wk     = (const float*)d_in[5];
    const float* bk     = (const float*)d_in[6];
    const float* Wv     = (const float*)d_in[7];
    const float* bv     = (const float*)d_in[8];
    const float* Wp     = (const float*)d_in[9];
    const float* bp     = (const float*)d_in[10];

    float* y_out = (float*)d_out;            // [16384,1024] fp32
    float* pk    = y_out + 16777216;         // present[0]=k, [B,H,T,D] fp32
    float* pv    = pk + 16777216;            // present[1]=v

    char* ws = (char*)d_ws;
    unsigned short* A0  = (unsigned short*)(ws);               // 8   MB im2col
    unsigned short* Wc  = (unsigned short*)(ws + 8388608);     // 0.5 MB conv_w bf16 [N,K]
    unsigned short* Wqb = (unsigned short*)(ws + 8912896);     // 2 MB  Wq^T bf16 [N,K]
    unsigned short* Wkb = (unsigned short*)(ws + 11010048);
    unsigned short* Wvb = (unsigned short*)(ws + 13107200);
    unsigned short* Wpb = (unsigned short*)(ws + 15204352);
    unsigned short* tok = (unsigned short*)(ws + 17301504);    // 32 MB tokens bf16
    unsigned short* Qbf = (unsigned short*)(ws + 50855936);    // 32 MB q bf16
    unsigned short* Ybf = (unsigned short*)(ws + 84410368);    // 32 MB attn out bf16

    // Staging: im2col + weight convert/transpose (all weights become [N,K] bf16)
    k_im2col<<<16384, 256, 0, stream>>>(x, A0);
    k_cvt<<<1024, 256, 0, stream>>>(conv_w, Wc, 262144);
    dim3 tg(32, 32), tb(32, 8);
    k_wt<<<tg, tb, 0, stream>>>(Wq, Wqb);
    k_wt<<<tg, tb, 0, stream>>>(Wk, Wkb);
    k_wt<<<tg, tb, 0, stream>>>(Wv, Wvb);
    k_wt<<<tg, tb, 0, stream>>>(Wp, Wpb);

    dim3 g(8, 128);   // N/128 x M/128
    // Patch embed: tok = im2col(x) @ conv_w^T + conv_b (bf16 out)
    k_gemm<<<g, 256, 0, stream>>>(A0, Wc, conv_b, nullptr, 0, tok, 16384, 1024, 256);
    // Q (bf16 for attention)
    k_gemm<<<g, 256, 0, stream>>>(tok, Wqb, bq, nullptr, 0, Qbf, 16384, 1024, 1024);
    // K, V: fp32 straight into present ([B,H,T,D] permute)
    k_gemm<<<g, 256, 0, stream>>>(tok, Wkb, bk, pk, 2, nullptr, 16384, 1024, 1024);
    k_gemm<<<g, 256, 0, stream>>>(tok, Wvb, bv, pv, 2, nullptr, 16384, 1024, 1024);
    // Attention (reads fp32 K/V from present, converts while staging to LDS)
    k_attn<<<512, 512, 157696, stream>>>(pk, pv, Qbf, Ybf);
    // Output projection -> y (fp32)
    k_gemm<<<g, 256, 0, stream>>>(Ybf, Wpb, bp, y_out, 1, nullptr, 16384, 1024, 1024);
}
